// NearestEmbed_39307540693106
// MI455X (gfx1250) — matrix operand order repaired and checked by reading, not compiled, not run on hardware
//
#include <hip/hip_runtime.h>

#define D 256

typedef __attribute__((ext_vector_type(16))) __bf16 v16bf;
typedef __attribute__((ext_vector_type(8)))  float  v8f;

union FragU { uint4 q[2]; v16bf v; };

// ---- fp32 -> (hi, lo) bf16 split helpers (RNE) ----
__device__ __forceinline__ unsigned short f2bf(float f) {
    unsigned int u = __float_as_uint(f);
    unsigned int r = (u + 0x7FFFu + ((u >> 16) & 1u)) >> 16;
    return (unsigned short)r;
}
__device__ __forceinline__ float bf2f(unsigned short b) {
    return __uint_as_float(((unsigned int)b) << 16);
}

__global__ void split_bf16_kernel(const float* __restrict__ src,
                                  unsigned short* __restrict__ hi,
                                  unsigned short* __restrict__ lo, int n) {
    int i = blockIdx.x * blockDim.x + threadIdx.x;
    int stride = gridDim.x * blockDim.x;
    for (; i < n; i += stride) {
        float v = src[i];
        unsigned short h = f2bf(v);
        float hf = bf2f(h);
        hi[i] = h;
        lo[i] = f2bf(v - hf);
    }
}

// one wave per codeword: wn[k] = ||w_k||^2
__global__ void wnorm_kernel(const float* __restrict__ w,
                             float* __restrict__ wn, int K) {
    int wave = (blockIdx.x * blockDim.x + threadIdx.x) >> 5;
    int lane = threadIdx.x & 31;
    if (wave >= K) return;
    const float* row = w + (size_t)wave * D;
    float s = 0.f;
    #pragma unroll
    for (int i = 0; i < D / 32; ++i) {
        float v = row[lane + 32 * i];
        s += v * v;
    }
    #pragma unroll
    for (int m = 16; m >= 1; m >>= 1) s += __shfl_xor(s, m, 32);
    if (lane == 0) wn[wave] = s;
}

__global__ __launch_bounds__(256)
void nearest_embed_kernel(const unsigned short* __restrict__ xh,
                          const unsigned short* __restrict__ xl,
                          const unsigned short* __restrict__ wh,
                          const unsigned short* __restrict__ wl,
                          const float* __restrict__ wnorm,
                          const float* __restrict__ weight,
                          float* __restrict__ outq,
                          float* __restrict__ outidx,
                          int K) {
    __shared__ float sD[8][16];
    __shared__ int   sI[8][16];
    __shared__ int   sBest[16];

    const int tid  = threadIdx.x;
    const int wave = tid >> 5;     // 0..7
    const int lane = tid & 31;
    const int g    = lane >> 4;    // 16-lane half
    const int l16  = lane & 15;
    const int rowBase = blockIdx.x * 16;

    // ---- A fragments: 16 rows x 256 k, hi & lo, resident in VGPRs ----
    // ISA 16-bit A 16x32 layout: lane holds row M = lane%16;
    // VGPR0-3 = K = 8g..8g+7, VGPR4-7 = K = 16+8g..16+8g+7.
    FragU Ah[8], Al[8];
    {
        const size_t rowOff = (size_t)(rowBase + l16) * D;
        #pragma unroll
        for (int ks = 0; ks < 8; ++ks) {
            const int kb = ks * 32 + g * 8;
            Ah[ks].q[0] = *(const uint4*)(xh + rowOff + kb);
            Ah[ks].q[1] = *(const uint4*)(xh + rowOff + kb + 16);
            Al[ks].q[0] = *(const uint4*)(xl + rowOff + kb);
            Al[ks].q[1] = *(const uint4*)(xl + rowOff + kb + 16);
        }
    }

    float best[8];
    int   bidx[8];
    #pragma unroll
    for (int v = 0; v < 8; ++v) { best[v] = 3.4e38f; bidx[v] = 0x7fffffff; }

    // ---- sweep codebook: 8 waves stripe 16-wide N tiles ----
    for (int nt = wave * 16; nt < K; nt += 16 * 8) {
        // B layout: lane holds col N = nt + lane%16; VGPR0-3 = K = 8g..8g+7, VGPR4-7 = K = 16+8g..16+8g+7
        const size_t nOff = (size_t)(nt + l16) * D + (size_t)g * 8;
        v8f c = {};
        #pragma unroll
        for (int ks = 0; ks < 8; ++ks) {
            FragU Bh, Bl;
            const unsigned short* ph = wh + nOff + ks * 32;
            const unsigned short* pl = wl + nOff + ks * 32;
            Bh.q[0] = *(const uint4*)(ph);
            Bh.q[1] = *(const uint4*)(ph + 16);
            Bl.q[0] = *(const uint4*)(pl);
            Bl.q[1] = *(const uint4*)(pl + 16);
            // dot ~= xh*wh + xl*wh + xh*wl  (fp32-accurate split product)
            c = __builtin_amdgcn_wmma_f32_16x16x32_bf16(false, Ah[ks].v, false, Bh.v,
                                                        (short)0, c, false, false);
            c = __builtin_amdgcn_wmma_f32_16x16x32_bf16(false, Al[ks].v, false, Bh.v,
                                                        (short)0, c, false, false);
            c = __builtin_amdgcn_wmma_f32_16x16x32_bf16(false, Ah[ks].v, false, Bl.v,
                                                        (short)0, c, false, false);
        }
        const float wn = wnorm[nt + l16];
        const int   ni = nt + l16;
        #pragma unroll
        for (int v = 0; v < 8; ++v) {
            float d = wn - 2.0f * c[v];   // ||w||^2 - 2 x.w  (||x||^2 constant)
            if (d < best[v] || (d == best[v] && ni < bidx[v])) {
                best[v] = d; bidx[v] = ni;
            }
        }
    }

    // ---- argmin reduce across the 16 lanes of each half (C layout: M = v + 8g) ----
    #pragma unroll
    for (int v = 0; v < 8; ++v) {
        float d = best[v]; int bi = bidx[v];
        #pragma unroll
        for (int m = 8; m >= 1; m >>= 1) {
            float od = __shfl_xor(d, m, 32);
            int   oi = __shfl_xor(bi, m, 32);
            if (od < d || (od == d && oi < bi)) { d = od; bi = oi; }
        }
        if (l16 == 0) {
            int row = v + 8 * g;
            sD[wave][row] = d;
            sI[wave][row] = bi;
        }
    }
    __syncthreads();

    // ---- reduce across the 8 waves ----
    if (tid < 16) {
        float d = sD[0][tid]; int bi = sI[0][tid];
        #pragma unroll
        for (int w2 = 1; w2 < 8; ++w2) {
            float od = sD[w2][tid]; int oi = sI[w2][tid];
            if (od < d || (od == d && oi < bi)) { d = od; bi = oi; }
        }
        sBest[tid] = bi;
        outidx[rowBase + tid] = (float)bi;
    }
    __syncthreads();

    // ---- gather quantized rows (coalesced: tid = column) ----
    for (int r = 0; r < 16; ++r) {
        int bi = sBest[r];
        outq[(size_t)(rowBase + r) * D + tid] = weight[(size_t)bi * D + tid];
    }
}

extern "C" void kernel_launch(void* const* d_in, const int* in_sizes, int n_in,
                              void* d_out, int out_size, void* d_ws, size_t ws_size,
                              hipStream_t stream) {
    const float* x = (const float*)d_in[0];   // (N, 256) fp32
    const float* w = (const float*)d_in[1];   // (K, 256) fp32
    const int N = in_sizes[0] / D;            // 16384
    const int K = in_sizes[1] / D;            // 8192

    // workspace layout: xh, xl (N*D bf16 each), wh, wl (K*D bf16 each), wnorm (K fp32)
    unsigned short* xh = (unsigned short*)d_ws;
    unsigned short* xl = xh + (size_t)N * D;
    unsigned short* wh = xl + (size_t)N * D;
    unsigned short* wl = wh + (size_t)K * D;
    float*          wn = (float*)(wl + (size_t)K * D);

    float* outq   = (float*)d_out;            // N*D quantized fp32
    float* outidx = outq + (size_t)N * D;     // N indices

    split_bf16_kernel<<<1024, 256, 0, stream>>>(x, xh, xl, N * D);
    split_bf16_kernel<<<512, 256, 0, stream>>>(w, wh, wl, K * D);
    wnorm_kernel<<<(K * 32 + 255) / 256, 256, 0, stream>>>(w, wn, K);
    nearest_embed_kernel<<<N / 16, 256, 0, stream>>>(xh, xl, wh, wl, wn, w,
                                                     outq, outidx, K);
}
